// PredictorLG_10471130268343
// MI455X (gfx1250) — compile-verified, matrix operand order checked by reading, not run on hardware
//
#include <hip/hip_runtime.h>
#include <cstdint>
#include <cstddef>

// ---------- CDNA5 WMMA / TDM types ----------
typedef __attribute__((ext_vector_type(16))) __bf16        v16bf;
typedef __attribute__((ext_vector_type(8)))  float         v8f;
typedef __attribute__((ext_vector_type(8)))  unsigned int  v8u;
typedef __attribute__((ext_vector_type(4)))  unsigned int  tdm_g0_t;
typedef __attribute__((ext_vector_type(8)))  int           tdm_g1_t;
typedef __attribute__((ext_vector_type(4)))  int           tdm_g2_t;

typedef unsigned short u16;
typedef unsigned int   u32;

#ifndef USE_TDM
#define USE_TDM 1
#endif

#define BATCH   64
#define C0      384
#define HW      1024
#define MROWS   65536            // B*H*W
#define OUT_X   25165824         // 64*384*32*32
#define OUT_I1  25165824
#define OUT_I2  (25165824 + 32768)

__device__ __forceinline__ u16 f2bf(float f) {
    u32 u = __builtin_bit_cast(u32, f);
    u += 0x7FFFu + ((u >> 16) & 1u);               // round-to-nearest-even
    return (u16)(u >> 16);
}
__device__ __forceinline__ float bf2f(u16 h) {
    u32 u = ((u32)h) << 16;
    return __builtin_bit_cast(float, u);
}
__device__ __forceinline__ float gelu_exact(float v) {
    return 0.5f * v * (1.0f + erff(v * 0.7071067811865475f));
}

// ---------- TDM: 1-D contiguous tile Global -> LDS via Tensor Data Mover.
// D# per CDNA5 ISA: group0 {count=1, lds_addr,
// global_addr[56:0], type=2}; group1 {wg_mask=0, data_size=3 (8B units),
// tensor_dim0 = tile_dim0 = bytes/8, tensor_dim1=1, tile_dim1/2 unused}.
// Groups 2/3 zero (tile dims 2+ unused). Tracked with TENSORcnt. ----------
__device__ __forceinline__ void tdm_load_1d(const void* gptr, u32 lds_off, u32 bytes) {
    u32 units = bytes >> 3;                                   // 8-byte elements
    unsigned long long ga = (unsigned long long)gptr;
    tdm_g0_t g0;
    g0[0] = 1u;                                               // count=1, user mode
    g0[1] = lds_off;                                          // lds_addr (bytes)
    g0[2] = (u32)ga;                                          // global_addr[31:0]
    g0[3] = (u32)((ga >> 32) & 0x01FFFFFFu) | 0x80000000u;    // global_addr[56:32] | type=2
    tdm_g1_t g1;
    g1[0] = (int)(3u << 16);                                  // wg_mask=0, data_size=3 (8B)
    g1[1] = (int)((units & 0xFFFFu) << 16);                   // tensor_dim0[15:0] @ bits 63:48
    g1[2] = (int)(((units >> 16) & 0xFFFFu) | (1u << 16));    // tensor_dim0[31:16], tensor_dim1=1
    g1[3] = (int)((units & 0xFFFFu) << 16);                   // tile_dim0 @ bits 127:112
    g1[4] = 0;                                                // tile_dim1=0, tile_dim2=0
    g1[5] = (int)units;                                       // tensor_dim0_stride[31:0]
    g1[6] = 0;
    g1[7] = 0;
    tdm_g2_t gz = { 0, 0, 0, 0 };
#if __clang_major__ >= 23
    tdm_g1_t z8 = { 0, 0, 0, 0, 0, 0, 0, 0 };
    __builtin_amdgcn_tensor_load_to_lds(g0, g1, gz, gz, z8, 0);
#else
    __builtin_amdgcn_tensor_load_to_lds(g0, g1, gz, gz, 0);
#endif
}

// ---------- Stage 0: pack W[Cout][Cin] into WMMA B-fragment lane layout:
// panel[(ntile*KK + kk)*32 + lane][16 bf16] -> one contiguous 32B load/lane.
// K-striping mirrors ISA 16-bit A 16x32 layout. ----------
__global__ void pack_weights_kernel(const float* __restrict__ W, u16* __restrict__ Bp,
                                    int Cin, int Cout) {
    int total = (Cout >> 4) * (Cin >> 5) * 512;
    int t = blockIdx.x * blockDim.x + threadIdx.x;
    if (t >= total) return;
    int e    = t & 15;
    int lane = (t >> 4) & 31;
    int kk   = (t >> 9) % (Cin >> 5);
    int nt   = t / ((Cin >> 5) << 9);
    int nl = lane & 15, hi = lane >> 4;
    int pe = e >> 1;
    int K  = ((pe < 4) ? 0 : 16) + hi * 8 + (pe & 3) * 2 + (e & 1);
    int c  = (kk << 5) + K;
    int o  = (nt << 4) + nl;
    Bp[t]  = f2bf(W[(size_t)o * Cin + c]);
}

// ---------- Stage 1: channel LayerNorm at each (b,h,w), row-major bf16 out ----------
__global__ __launch_bounds__(256)
void ln_pack_kernel(const float* __restrict__ x, const float* __restrict__ lnw,
                    const float* __restrict__ lnb, u32* __restrict__ xn) {
    int p  = blockIdx.x * blockDim.x + threadIdx.x;     // 0..65535
    int b  = p >> 10, hw = p & 1023;
    const float* base = x + (size_t)b * C0 * HW + hw;
    float s = 0.f, s2 = 0.f;
    for (int c = 0; c < C0; ++c) { float v = base[(size_t)c * HW]; s += v; s2 += v * v; }
    float u    = s  * (1.0f / C0);
    float var  = s2 * (1.0f / C0) - u * u;
    float rinv = rsqrtf(var + 1e-6f);
    u32* dst = xn + (size_t)p * (C0 / 2);
    for (int c2 = 0; c2 < C0 / 2; ++c2) {
        int c = c2 * 2;
        float v0 = (base[(size_t)c * HW]       - u) * rinv * lnw[c]     + lnb[c];
        float v1 = (base[(size_t)(c + 1) * HW] - u) * rinv * lnw[c + 1] + lnb[c + 1];
        dst[c2] = (u32)f2bf(v0) | ((u32)f2bf(v1) << 16);
    }
}

// ---------- WMMA GEMM: Y[p][n] = gelu(A[p][:] . W[n][:] + bias[n])
// 256 threads = 8 wave32; block owns 64 rows. A(64 x CIN bf16, <=48KB) staged
// in LDS once via TDM (wave0 issues tensor_load_to_lds + s_wait_tensorcnt 0,
// barrier releases the rest), reused across all N tiles. K loop fully unrolled
// -> back-to-back v_wmma_f32_16x16x32_bf16; B loads use a hoisted per-tile
// base with constant 1KB K-step offsets (folded into the 24-bit IOFFSET).
// MODE 0: row-major out. MODE 1: n<192 row-major into X3; n>=192 column-major
// into Yg (one b128 store per lane) for the deterministic global-mean. ----------
template<int CIN, int COUT, int MODE>
__global__ __launch_bounds__(256)
void gemm_wmma_kernel(const u16* __restrict__ A, const u16* __restrict__ Bp,
                      const float* __restrict__ bias,
                      u16* __restrict__ outRM, int ldo, u16* __restrict__ outCM) {
    __shared__ __align__(16) u16 lA[64 * CIN];
    const int tid = threadIdx.x;
    const int p0  = blockIdx.x * 64;
#if USE_TDM
    if (tid == 0) {
        tdm_load_1d(A + (size_t)p0 * CIN, (u32)(unsigned long long)(void*)lA, 64 * CIN * 2);
        __builtin_amdgcn_s_wait_tensorcnt(0);
    }
    __syncthreads();
#else
    {   // cooperative coalesced A-panel load -> LDS
        const uint4* src = (const uint4*)(A + (size_t)p0 * CIN);
        uint4* dst = (uint4*)lA;
        for (int i = tid; i < (64 * CIN) / 8; i += 256) dst[i] = src[i];
    }
    __syncthreads();
#endif

    const int wave = tid >> 5, lane = tid & 31;
    const int mi = wave & 3, nh = wave >> 2;
    const int nl = lane & 15, hi = lane >> 4;
    const int rowLocal = mi * 16 + nl;              // A-frag: m = lane&15
    const u32* lAu = (const u32*)lA;
    constexpr int KK = CIN >> 5;

    for (int t = 0; t < (COUT >> 5); ++t) {
        const int n0 = t * 32 + nh * 16;
        const int n  = n0 + nl;
        const float bv = bias[n];
        v8f acc;
        #pragma unroll
        for (int r = 0; r < 8; ++r) acc[r] = bv;

        // per-tile B base; K steps are +512 elements (1024B) constant offsets
        const u16* bbase = Bp + ((size_t)(n0 >> 4) * KK * 32 + lane) * 16;

        #pragma unroll
        for (int kk = 0; kk < KK; ++kk) {
            v8u au;
            const int rb = rowLocal * CIN + kk * 32;
            #pragma unroll
            for (int pe = 0; pe < 8; ++pe) {        // 8x ds_load_b32, pairs (K,K+1)
                int kbase = ((pe < 4) ? 0 : 16) + hi * 8 + (pe & 3) * 2;
                au[pe] = lAu[(rb + kbase) >> 1];
            }
            v16bf afrag = __builtin_bit_cast(v16bf, au);
            v16bf bfrag = __builtin_bit_cast(v16bf, *(const v8u*)(bbase + kk * 512));
            acc = __builtin_amdgcn_wmma_f32_16x16x32_bf16(false, afrag, false, bfrag,
                                                          (short)0, acc, false, false);
        }

        if (MODE == 0 || n < 192) {
            #pragma unroll
            for (int r = 0; r < 8; ++r) {           // D layout: M = r + 8*hi, N = lane&15
                float v = gelu_exact(acc[r]);
                int p = p0 + mi * 16 + hi * 8 + r;
                outRM[(size_t)p * ldo + n] = f2bf(v);
            }
        } else {                                    // global half -> column-major Yg
            u32 w[4];
            #pragma unroll
            for (int j = 0; j < 4; ++j)
                w[j] = (u32)f2bf(gelu_exact(acc[2 * j])) |
                       ((u32)f2bf(gelu_exact(acc[2 * j + 1])) << 16);
            int pbase = p0 + mi * 16 + hi * 8;      // 8 consecutive rows per lane
            *(uint4*)(outCM + (size_t)(n - 192) * MROWS + pbase) = make_uint4(w[0], w[1], w[2], w[3]);
        }
    }
}

// ---------- Stage 2b: deterministic mean over H*W per (b, global-channel) ----------
__global__ __launch_bounds__(256)
void reduce_global_kernel(const u16* __restrict__ Yg, float* __restrict__ gmean) {
    int gw   = blockIdx.x * (blockDim.x >> 5) + (threadIdx.x >> 5);  // 0..12287
    int lane = threadIdx.x & 31;
    int j = gw % 192, b = gw / 192;
    const u16* src = Yg + (size_t)j * MROWS + b * 1024;
    float s = 0.f;
    for (int i = lane; i < 1024; i += 32) s += bf2f(src[i]);
    for (int off = 16; off > 0; off >>= 1) s += __shfl_down(s, off, 32);
    if (lane == 0) gmean[b * 192 + j] = s * (1.0f / 1024.0f);
}

__global__ __launch_bounds__(256)
void broadcast_global_kernel(const float* __restrict__ gmean, u32* __restrict__ X3u) {
    int t = blockIdx.x * blockDim.x + threadIdx.x;  // 65536*96
    int p = t / 96, jj = t % 96, b = p >> 10;
    u32 w = (u32)f2bf(gmean[b * 192 + 2 * jj]) | ((u32)f2bf(gmean[b * 192 + 2 * jj + 1]) << 16);
    X3u[(size_t)p * 192 + 96 + jj] = w;             // channels 192..383 of X3 row p
}

// ---------- Stage 5: 96->2 matvec + log_softmax, keep class-0 log-prob ----------
__global__ __launch_bounds__(256)
void score_kernel(const u16* __restrict__ X5, const float* __restrict__ w3,
                  const float* __restrict__ b3, float* __restrict__ score) {
    int p = blockIdx.x * blockDim.x + threadIdx.x;
    const u16* row = X5 + (size_t)p * 96;
    float z0 = b3[0], z1 = b3[1];
    for (int c = 0; c < 96; ++c) {
        float v = bf2f(row[c]);
        z0 += v * w3[c];
        z1 += v * w3[96 + c];
    }
    float m = fmaxf(z0, z1);
    score[p] = z0 - (m + logf(expf(z0 - m) + expf(z1 - m)));
}

// ---------- Stage 6: per-batch descending bitonic argsort of 1024 scores ----------
__global__ __launch_bounds__(512)
void sort_kernel(const float* __restrict__ score, float* __restrict__ out) {
    __shared__ float s[1024];
    __shared__ int  id[1024];
    int b = blockIdx.x, tid = threadIdx.x;
    for (int i = tid; i < 1024; i += 512) { s[i] = score[b * 1024 + i]; id[i] = i; }
    __syncthreads();
    for (int k = 2; k <= 1024; k <<= 1) {
        for (int j = k >> 1; j > 0; j >>= 1) {
            for (int t = tid; t < 1024; t += 512) {
                int ixj = t ^ j;
                if (ixj > t) {
                    bool desc = ((t & k) == 0);
                    bool sw = desc ? (s[t] < s[ixj]) : (s[t] > s[ixj]);
                    if (sw) {
                        float tf = s[t]; s[t] = s[ixj]; s[ixj] = tf;
                        int  ti = id[t]; id[t] = id[ixj]; id[ixj] = ti;
                    }
                }
            }
            __syncthreads();
        }
    }
    out[OUT_I1 + b * 512 + tid] = (float)id[tid];
    out[OUT_I2 + b * 512 + tid] = (float)id[tid + 512];
}

extern "C" void kernel_launch(void* const* d_in, const int* in_sizes, int n_in,
                              void* d_out, int out_size, void* d_ws, size_t ws_size,
                              hipStream_t stream) {
    const float* x    = (const float*)d_in[0];
    const float* lnw  = (const float*)d_in[1];
    const float* lnb  = (const float*)d_in[2];
    const float* in_w = (const float*)d_in[3];
    const float* in_b = (const float*)d_in[4];
    const float* w1   = (const float*)d_in[5];
    const float* b1   = (const float*)d_in[6];
    const float* w2   = (const float*)d_in[7];
    const float* b2   = (const float*)d_in[8];
    const float* w3   = (const float*)d_in[9];
    const float* b3   = (const float*)d_in[10];
    float* out = (float*)d_out;
    char*  ws  = (char*)d_ws;

    size_t off = 0;
    auto carve = [&](size_t bytes) { size_t o = off; off += (bytes + 255) & ~(size_t)255; return o; };
    size_t oXn  = carve((size_t)MROWS * 384 * 2);   // stage-2 A panel (bf16)
    size_t oX3  = carve((size_t)MROWS * 384 * 2);   // stage-3 input (local | broadcast)
    size_t oYg  = carve((size_t)MROWS * 192 * 2);   // global half, column-major
    size_t oGm  = carve(64 * 192 * 4);
    size_t oSc  = carve((size_t)MROWS * 4);
    size_t oBp0 = carve(384 * 384 * 2);
    size_t oBp1 = carve(192 * 384 * 2);
    size_t oBp2 = carve(96 * 192 * 2);
    size_t oX4  = oXn;                              // Xn dead after stage 2: alias
    size_t oX5  = oXn + (size_t)MROWS * 192 * 2;

    // pass through input_x (100 MB D2D, graph-capturable)
    hipMemcpyAsync(out, x, (size_t)OUT_X * 4, hipMemcpyDeviceToDevice, stream);

    pack_weights_kernel<<<(24 * 12 * 512 + 255) / 256, 256, 0, stream>>>(in_w, (u16*)(ws + oBp0), 384, 384);
    pack_weights_kernel<<<(12 * 12 * 512 + 255) / 256, 256, 0, stream>>>(w1,   (u16*)(ws + oBp1), 384, 192);
    pack_weights_kernel<<<( 6 *  6 * 512 + 255) / 256, 256, 0, stream>>>(w2,   (u16*)(ws + oBp2), 192,  96);

    ln_pack_kernel<<<MROWS / 256, 256, 0, stream>>>(x, lnw, lnb, (u32*)(ws + oXn));

    gemm_wmma_kernel<384, 384, 1><<<MROWS / 64, 256, 0, stream>>>(
        (u16*)(ws + oXn), (u16*)(ws + oBp0), in_b, (u16*)(ws + oX3), 384, (u16*)(ws + oYg));

    reduce_global_kernel<<<(64 * 192) / 8, 256, 0, stream>>>((u16*)(ws + oYg), (float*)(ws + oGm));
    broadcast_global_kernel<<<(MROWS * 96) / 256, 256, 0, stream>>>((float*)(ws + oGm), (u32*)(ws + oX3));

    gemm_wmma_kernel<384, 192, 0><<<MROWS / 64, 256, 0, stream>>>(
        (u16*)(ws + oX3), (u16*)(ws + oBp1), b1, (u16*)(ws + oX4), 192, nullptr);
    gemm_wmma_kernel<192, 96, 0><<<MROWS / 64, 256, 0, stream>>>(
        (u16*)(ws + oX4), (u16*)(ws + oBp2), b2, (u16*)(ws + oX5), 96, nullptr);

    score_kernel<<<MROWS / 256, 256, 0, stream>>>((u16*)(ws + oX5), w3, b3, (float*)(ws + oSc));
    sort_kernel<<<64, 512, 0, stream>>>((float*)(ws + oSc), out);
}
